// AttentionConv_41961830482236
// MI455X (gfx1250) — matrix operand order repaired and checked by reading, not compiled, not run on hardware
//
#include <hip/hip_runtime.h>
#include <hip/hip_bf16.h>

// ---------------------------------------------------------------------------
// AttentionConv (K=7 window local attention, C=64, G=8) for gfx1250 (MI455X)
//   Phase 1: QKV 1x1-conv projections as WMMA f16 GEMMs (f32 accumulate)
//   Phase 2: per-channel 7x7 windowed softmax-attention; K/V halo tiles staged
//            into LDS via GLOBAL_LOAD_ASYNC_TO_LDS_B32 (ASYNCcnt path) when
//            the toolchain exposes the builtin, else synchronous fallback.
// ---------------------------------------------------------------------------

typedef __attribute__((ext_vector_type(16))) _Float16 v16h;
typedef __attribute__((ext_vector_type(8)))  float    v8f;

#define CCH   64
#define HH    56
#define WW    56
#define HWSZ  (HH * WW)          // 3136
#define NB    4
#define NPIX  (NB * HWSZ)        // 12544
#define KWIN  7
#define PADR  3
#define TILE  16
#define HALO  (TILE + 2 * PADR)  // 22

#if defined(__has_builtin)
#  if __has_builtin(__builtin_amdgcn_global_load_async_to_lds_b32)
#    define USE_ASYNC_LDS 1
#  endif
#endif

typedef __attribute__((address_space(1))) int g_int;   // global (AS1) int
typedef __attribute__((address_space(3))) int l_int;   // LDS (AS3) int

__device__ __forceinline__ void wait_asynccnt0() {
#if defined(__has_builtin)
#  if __has_builtin(__builtin_amdgcn_s_wait_asynccnt)
    __builtin_amdgcn_s_wait_asynccnt(0);
    return;
#  endif
#endif
    asm volatile("s_wait_asynccnt 0x0" ::: "memory");
}

// ---------------------------------------------------------------------------
// Kernel 1: q/k/v = W{q,k,v} @ x  as 16x16 WMMA tiles.
// Grid.x = NPIX/16 = 784 pixel tiles; block = 256 threads (8 waves).
// Each wave builds the shared B operand (x tile, 64xK) once, then loops over
// the 12 (matrix, m-tile) tasks doing 2x v_wmma_f32_16x16x32_f16 each.
// ---------------------------------------------------------------------------
__global__ __launch_bounds__(256)
void qkv_wmma_kernel(const float* __restrict__ x,
                     const float* __restrict__ Wq,
                     const float* __restrict__ Wk,
                     const float* __restrict__ Wv,
                     float* __restrict__ qo,
                     float* __restrict__ ko,
                     float* __restrict__ vo)
{
    const int ptile = blockIdx.x;            // 0..783 (batch-aligned: 3136%16==0)
    const int lane  = threadIdx.x & 31;
    const int wave  = threadIdx.x >> 5;

    // Pull the (tiny, hot) weight matrices toward the caches early.
    __builtin_prefetch(Wq + lane * 64, 0, 0);   // -> global_prefetch_b8
    __builtin_prefetch(Wk + lane * 64, 0, 0);
    __builtin_prefetch(Wv + lane * 64, 0, 0);

    const int col   = lane & 15;             // matrix column within tile
    const int hsel  = lane >> 4;             // K-half selector within lane group

    const int gp    = ptile * 16 + col;      // global pixel (batch-major)
    const int b     = gp / HWSZ;
    const int hw    = gp - b * HWSZ;
    const float* xb = x + (size_t)b * CCH * HWSZ + hw;

    // B operand (x tile), B 32x16, contiguous k order used by this code (not AMD's table order; A matches):
    //   lane: N = lane&15 (pixel), VGPR j element e holds K = 16*hsel + 2j + e
    // bx0 covers K=0..31, bx1 covers K=32..63.
    v16h bx0, bx1;
#pragma unroll
    for (int j = 0; j < 8; ++j) {
        const int kk = hsel * 16 + 2 * j;
        bx0[2 * j + 0] = (_Float16)xb[(size_t)(kk + 0)  * HWSZ];
        bx0[2 * j + 1] = (_Float16)xb[(size_t)(kk + 1)  * HWSZ];
        bx1[2 * j + 0] = (_Float16)xb[(size_t)(kk + 32) * HWSZ];
        bx1[2 * j + 1] = (_Float16)xb[(size_t)(kk + 33) * HWSZ];
    }

    const float* Wmats[3] = { Wq, Wk, Wv };
    float*       Outs [3] = { qo, ko, vo };

    for (int t = wave; t < 12; t += 8) {
        const int m  = t >> 2;               // which matrix (q/k/v)
        const int mt = t & 3;                // which 16-row out-channel tile
        const float* Wm = Wmats[m];
        const int row = mt * 16 + (lane & 15);

        // A operand (weights), A 16x32, same contiguous k order as B (not AMD's table order):
        //   lane: M = lane&15, VGPR j element e holds
        //   K = 16*hsel + 2j + e
        v16h a0, a1;
#pragma unroll
        for (int j = 0; j < 8; ++j) {
            const int kb = (hsel << 4) + (j << 1);
            const float* wr = Wm + (size_t)row * CCH + kb;
            a0[2 * j + 0] = (_Float16)wr[0];
            a0[2 * j + 1] = (_Float16)wr[1];
            a1[2 * j + 0] = (_Float16)wr[32];
            a1[2 * j + 1] = (_Float16)wr[33];
        }

        v8f acc = {};
        acc = __builtin_amdgcn_wmma_f32_16x16x32_f16(
                  false, a0, false, bx0, (short)0, acc, false, false);
        acc = __builtin_amdgcn_wmma_f32_16x16x32_f16(
                  false, a1, false, bx1, (short)0, acc, false, false);

        // D layout: v8f element r -> out row = mt*16 + r + 8*hsel, col = lane&15
        float* outp = Outs[m] + (size_t)b * CCH * HWSZ + hw;
#pragma unroll
        for (int r = 0; r < 8; ++r) {
            const int orow = mt * 16 + r + 8 * hsel;
            outp[(size_t)orow * HWSZ] = acc[r];
        }
    }
}

// ---------------------------------------------------------------------------
// Kernel 2: windowed softmax attention.
// grid = (16 spatial tiles, C=64, B=4); block = 256 threads (16x16 tile).
// LDS: 22x22 K tile, 22x22 V tile (zero-padded halo), 7 rel scalars.
// out[b,c,y,x] = sum_kk softmax_kk(q * k_win) * (v_win + rel)
// ---------------------------------------------------------------------------
__global__ __launch_bounds__(256)
void attn_window_kernel(const float* __restrict__ q,
                        const float* __restrict__ kin,
                        const float* __restrict__ vin,
                        const float* __restrict__ rel_h,
                        const float* __restrict__ rel_w,
                        float* __restrict__ out)
{
    __shared__ float sK[HALO * HALO];
    __shared__ float sV[HALO * HALO];
    __shared__ float sRel[KWIN];

    const int tid    = threadIdx.x;
    const int tileId = blockIdx.x;           // 0..15 (4x4 tiles of 16x16)
    const int c      = blockIdx.y;
    const int b      = blockIdx.z;
    const int ty0    = (tileId >> 2) * TILE;
    const int tx0    = (tileId & 3)  * TILE;

    const size_t plane = ((size_t)b * CCH + c) * HWSZ;
    const float* Kp = kin + plane;
    const float* Vp = vin + plane;

    // rel_h: (32,1,1,7,1) -> rel_h[c*7+dy] for c<32
    // rel_w: (32,1,1,1,7) -> rel_w[(c-32)*7+dx] for c>=32
    if (tid < KWIN) {
        sRel[tid] = (c < 32) ? rel_h[c * KWIN + tid]
                             : rel_w[(c - 32) * KWIN + tid];
    }

#if defined(USE_ASYNC_LDS)
    // Async staging: zero-prefill (halo padding), barrier to order DS vs ASYNC,
    // then EXEC-masked async copies for in-bounds elements, wait, barrier.
    for (int i = tid; i < HALO * HALO; i += 256) {
        sK[i] = 0.0f;
        sV[i] = 0.0f;
    }
    __syncthreads();
    for (int i = tid; i < HALO * HALO; i += 256) {
        const int ly = i / HALO;
        const int lx = i - ly * HALO;
        const int gy = ty0 + ly - PADR;
        const int gx = tx0 + lx - PADR;
        if ((gy >= 0) & (gy < HH) & (gx >= 0) & (gx < WW)) {
            const int gidx = gy * WW + gx;
            __builtin_amdgcn_global_load_async_to_lds_b32(
                (g_int*)(Kp + gidx), (l_int*)(&sK[i]), 0, 0);
            __builtin_amdgcn_global_load_async_to_lds_b32(
                (g_int*)(Vp + gidx), (l_int*)(&sV[i]), 0, 0);
        }
    }
    wait_asynccnt0();
    __syncthreads();
#else
    for (int i = tid; i < HALO * HALO; i += 256) {
        const int ly = i / HALO;
        const int lx = i - ly * HALO;
        const int gy = ty0 + ly - PADR;
        const int gx = tx0 + lx - PADR;
        const bool inb = (gy >= 0) & (gy < HH) & (gx >= 0) & (gx < WW);
        sK[i] = inb ? Kp[gy * WW + gx] : 0.0f;
        sV[i] = inb ? Vp[gy * WW + gx] : 0.0f;
    }
    __syncthreads();
#endif

    const int ly = tid >> 4;
    const int lx = tid & 15;
    const int gy = ty0 + ly;
    const int gx = tx0 + lx;
    if (gy >= HH || gx >= WW) return;

    const float qv = q[plane + gy * WW + gx];

    float s[KWIN * KWIN];
    float mx = -3.0e38f;
#pragma unroll
    for (int dy = 0; dy < KWIN; ++dy) {
#pragma unroll
        for (int dx = 0; dx < KWIN; ++dx) {
            const float sc = qv * sK[(ly + dy) * HALO + (lx + dx)];
            s[dy * KWIN + dx] = sc;
            mx = fmaxf(mx, sc);
        }
    }

    const bool isH = (c < 32);
    float denom = 0.0f;
    float num   = 0.0f;
#pragma unroll
    for (int dy = 0; dy < KWIN; ++dy) {
#pragma unroll
        for (int dx = 0; dx < KWIN; ++dx) {
            const float w   = __expf(s[dy * KWIN + dx] - mx);
            const float rel = isH ? sRel[dy] : sRel[dx];
            const float val = sV[(ly + dy) * HALO + (lx + dx)] + rel;
            denom += w;
            num   += w * val;
        }
    }

    out[plane + gy * WW + gx] = num / denom;
}

// ---------------------------------------------------------------------------
// Launch
// ---------------------------------------------------------------------------
extern "C" void kernel_launch(void* const* d_in, const int* in_sizes, int n_in,
                              void* d_out, int out_size, void* d_ws, size_t ws_size,
                              hipStream_t stream) {
    const float* x     = (const float*)d_in[0];
    const float* Wq    = (const float*)d_in[1];
    const float* Wk    = (const float*)d_in[2];
    const float* Wv    = (const float*)d_in[3];
    const float* rel_h = (const float*)d_in[4];
    const float* rel_w = (const float*)d_in[5];
    float* out = (float*)d_out;

    const size_t tsz = (size_t)NB * CCH * HWSZ;   // 802816 floats per tensor
    float* qs = (float*)d_ws;
    float* ks = qs + tsz;
    float* vs = ks + tsz;

    // Phase 1: QKV projection (784 pixel tiles x 8 waves)
    qkv_wmma_kernel<<<NPIX / 16, 256, 0, stream>>>(x, Wq, Wk, Wv, qs, ks, vs);

    // Phase 2: windowed attention (16 tiles x 64 channels x 4 batches)
    dim3 g2(16, CCH, NB);
    attn_window_kernel<<<g2, 256, 0, stream>>>(qs, ks, vs, rel_h, rel_w, out);
}
